// DeepseekV3MoE_73804718014884
// MI455X (gfx1250) — matrix operand order repaired and checked by reading, not compiled, not run on hardware
//
#include <hip/hip_runtime.h>
#include <hip/hip_bf16.h>

// DeepseekV3 MoE for gfx1250 (MI455X): bf16 WMMA (16x16x32), f32 accum.
// Pipeline: init -> gate/dispatch -> shared FFN1 -> shared FFN2 (writes y)
//           -> routed grouped FFN1 -> routed FFN2 + weighted atomic combine.
// Blocks cover 64 output rows (4 M-tiles): B fragment loaded once per K-step,
// reused across 4 WMMAs. All 4 A fragments hoisted (8x ds_load_b128, single
// dscnt wait) before 4 back-to-back WMMAs. A tiles double-buffered in LDS;
// B fragments loaded directly from global f32 + cvt_pk to bf16.

constexpr int   kH   = 2048;   // hidden
constexpr int   kT   = 2048;   // tokens (B*S)
constexpr int   kI   = 1408;   // routed intermediate
constexpr int   kE   = 16;     // experts
constexpr int   kK   = 6;      // top-k
constexpr int   kCap = 1536;   // capacity per expert
constexpr int   kSH  = 2816;   // shared intermediate (NSH*I)
constexpr float kRSF = 2.5f;
constexpr int   kLdA = 40;     // A tile leading dim (bf16): 80B rows, 16B-aligned
constexpr int   kBM  = 64;     // rows per block (4 M-tiles)

typedef __bf16 bf16;
typedef __attribute__((ext_vector_type(8)))  __bf16 v8bf;
typedef __attribute__((ext_vector_type(16))) __bf16 v16bf;
typedef __attribute__((ext_vector_type(8)))  float  v8f;

__device__ __forceinline__ v8f wmma_bf16(v16bf a, v16bf b, v8f c) {
  // (neg_a, A, neg_b, B, c_mod, C, reuse_a, reuse_b)
  return __builtin_amdgcn_wmma_f32_16x16x32_bf16(false, a, false, b,
                                                 (short)0, c, false, false);
}

// A fragment from a row-major 16x32 bf16 LDS tile (ld = kLdA).
// Lane (r, half) holds K = half*16+0..15 (one contiguous 32B run)
// -> two ds_load_b128.
__device__ __forceinline__ v16bf frag_a_lds(const bf16* As, int lane) {
  const int r = lane & 15;
  const int half = lane >> 4;
  const bf16* p = As + r * kLdA + half * 16;
  const v8bf lo = *(const v8bf*)(p);
  const v8bf hi = *(const v8bf*)(p + 8);
  return __builtin_shufflevector(lo, hi, 0, 1, 2, 3, 4, 5, 6, 7,
                                 8, 9, 10, 11, 12, 13, 14, 15);
}

// B fragment straight from global f32 weights: lane (c, half) holds
// K = half*16 + j (j=0..15) of column (ncol+c); col points at
// W[(k0 + half*16)*ldw + ncol + c].
__device__ __forceinline__ v16bf frag_b_gmem(const float* col, size_t ldw) {
  v16bf b;
#pragma unroll
  for (int j = 0; j < 16; ++j) b[j] = (bf16)col[(size_t)j * ldw];
  return b;
}

__device__ __forceinline__ v8bf pack8(float4 f0, float4 f1) {
  v8bf v;
  v[0] = (bf16)f0.x; v[1] = (bf16)f0.y; v[2] = (bf16)f0.z; v[3] = (bf16)f0.w;
  v[4] = (bf16)f1.x; v[5] = (bf16)f1.y; v[6] = (bf16)f1.z; v[7] = (bf16)f1.w;
  return v;
}

__device__ __forceinline__ float silu_f(float v) {
  return v / (1.0f + __expf(-v));
}

// 4 M-tiles x 1 N-tile WMMA step: hoist all A fragments, then 4 WMMAs.
__device__ __forceinline__ void mma_step4(const bf16* Ab, int lane,
                                          const v16bf& b, v8f acc[4]) {
  const v16bf a0 = frag_a_lds(Ab + 0 * 16 * kLdA, lane);
  const v16bf a1 = frag_a_lds(Ab + 1 * 16 * kLdA, lane);
  const v16bf a2 = frag_a_lds(Ab + 2 * 16 * kLdA, lane);
  const v16bf a3 = frag_a_lds(Ab + 3 * 16 * kLdA, lane);
  acc[0] = wmma_bf16(a0, b, acc[0]);
  acc[1] = wmma_bf16(a1, b, acc[1]);
  acc[2] = wmma_bf16(a2, b, acc[2]);
  acc[3] = wmma_bf16(a3, b, acc[3]);
}

// ---------------------------------------------------------------- init ----
__global__ void init_kernel(int* __restrict__ counts) {
  if (threadIdx.x < kE) counts[threadIdx.x] = 0;
}

// ----------------------------------------------------- gate + dispatch ----
__global__ void __launch_bounds__(64) gate_kernel(
    const float* __restrict__ x, const float* __restrict__ gw,
    const float* __restrict__ gb, int* __restrict__ counts,
    int* __restrict__ tok_ids, float* __restrict__ wslot) {
  const int t = blockIdx.x;
  const int tid = threadIdx.x;
  __shared__ float red[kE][65];

  float acc[kE];
#pragma unroll
  for (int e = 0; e < kE; ++e) acc[e] = 0.0f;
  const float* xp = x + (size_t)t * kH;
  for (int h = tid; h < kH; h += 64) {
    const float xv = xp[h];
#pragma unroll
    for (int e = 0; e < kE; ++e) acc[e] += xv * gw[e * kH + h];
  }
#pragma unroll
  for (int e = 0; e < kE; ++e) red[e][tid] = acc[e];
  __syncthreads();

  if (tid == 0) {
    float scores[kE], sc[kE];
    for (int e = 0; e < kE; ++e) {
      float s = 0.0f;
      for (int j = 0; j < 64; ++j) s += red[e][j];
      const float sig = 1.0f / (1.0f + __expf(-s));
      scores[e] = sig;
      sc[e] = sig + gb[e];
    }
    float gsc[4];
    for (int g = 0; g < 4; ++g) {
      float m1 = -1e30f, m2 = -1e30f;
      for (int j = 0; j < 4; ++j) {
        const float v = sc[4 * g + j];
        if (v > m1) { m2 = m1; m1 = v; } else if (v > m2) { m2 = v; }
      }
      gsc[g] = m1 + m2;
    }
    int g1 = 0;
    for (int g = 1; g < 4; ++g) if (gsc[g] > gsc[g1]) g1 = g;
    int g2 = -1;
    for (int g = 0; g < 4; ++g) {
      if (g == g1) continue;
      if (g2 < 0 || gsc[g] > gsc[g2]) g2 = g;
    }
    float tmp[kE];
    for (int e = 0; e < kE; ++e) {
      const int g = e >> 2;
      tmp[e] = (g == g1 || g == g2) ? sc[e] : 0.0f;
    }
    int idx[kK];
    float tw[kK];
    bool used[kE];
    for (int e = 0; e < kE; ++e) used[e] = false;
    float wsum = 0.0f;
    for (int k = 0; k < kK; ++k) {
      int best = 0;
      float bv = -1e30f;
      for (int e = 0; e < kE; ++e) {
        if (used[e]) continue;
        if (tmp[e] > bv) { bv = tmp[e]; best = e; }
      }
      used[best] = true;
      idx[k] = best;
      tw[k] = scores[best];
      wsum += tw[k];
    }
    const float inv = kRSF / (wsum + 1e-20f);
    for (int k = 0; k < kK; ++k) {
      const int e = idx[k];
      const int pos = atomicAdd(&counts[e], 1);
      if (pos < kCap) {
        tok_ids[e * kCap + pos] = t;
        wslot[e * kCap + pos] = tw[k] * inv;
      }
    }
  }
}

// -------------------------------------------------- shared expert FFN1 ----
__global__ void __launch_bounds__(256) shared_ffn1_kernel(
    const float* __restrict__ x, const float* __restrict__ wgp,
    const float* __restrict__ wup, bf16* __restrict__ hs) {
  const int t0 = blockIdx.y * kBM;
  const int wave = threadIdx.x >> 5;
  const int lane = threadIdx.x & 31;
  const int half = lane >> 4;
  const int nsub = wave & 3;
  const int ncol = blockIdx.x * 64 + nsub * 16;
  const float* __restrict__ W = (wave < 4) ? wgp : wup;

  __shared__ __align__(16) bf16 As[2][kBM * kLdA];
  __shared__ float Us[4][kBM][17];

  const int sr = threadIdx.x >> 2;        // 0..63 rows
  const int sc = (threadIdx.x & 3) * 8;   // 0,8,16,24 (floats)
  auto stageA = [&](int buf, int k0) {
    const float* src = &x[(size_t)(t0 + sr) * kH + k0 + sc];
    const float4 f0 = ((const float4*)src)[0];
    const float4 f1 = ((const float4*)src)[1];
    *(v8bf*)&As[buf][sr * kLdA + sc] = pack8(f0, f1);
  };

  stageA(0, 0);
  v8f acc[4] = {{}, {}, {}, {}};
  const float* wcol = W + (size_t)(half * 16) * kSH + ncol + (lane & 15);
  for (int k0 = 0, it = 0; k0 < kH; k0 += 32, ++it) {
    __syncthreads();
    if (k0 + 32 < kH) {
      stageA((it + 1) & 1, k0 + 32);
      __builtin_prefetch(wcol + (size_t)32 * kSH, 0, 0);
    }
    const v16bf b = frag_b_gmem(wcol, kSH);
    mma_step4(&As[it & 1][0], lane, b, acc);
    wcol += (size_t)32 * kSH;
  }

  const int c = lane & 15;
  const int rb = half * 8;
  if (wave >= 4) {
#pragma unroll
    for (int m = 0; m < 4; ++m)
#pragma unroll
      for (int r = 0; r < 8; ++r) Us[nsub][m * 16 + rb + r][c] = acc[m][r];
  }
  __syncthreads();
  if (wave < 4) {
#pragma unroll
    for (int m = 0; m < 4; ++m)
#pragma unroll
      for (int r = 0; r < 8; ++r) {
        const float g = acc[m][r];
        const float u = Us[nsub][m * 16 + rb + r][c];
        hs[(size_t)(t0 + m * 16 + rb + r) * kSH + ncol + c] =
            (bf16)(silu_f(g) * u);
      }
  }
}

// -------------------------------------------------- shared expert FFN2 ----
__global__ void __launch_bounds__(128) shared_ffn2_kernel(
    const bf16* __restrict__ hs, const float* __restrict__ wdn,
    float* __restrict__ y) {
  const int t0 = blockIdx.y * kBM;
  const int wave = threadIdx.x >> 5;
  const int lane = threadIdx.x & 31;
  const int half = lane >> 4;
  const int ncol = blockIdx.x * 64 + wave * 16;

  __shared__ __align__(16) bf16 As[2][kBM * kLdA];

  const int sr = threadIdx.x >> 1;         // 0..63 rows
  const int sc = (threadIdx.x & 1) * 16;   // 16-elem halves
  auto stageA = [&](int buf, int k0) {
    const bf16* src = &hs[(size_t)(t0 + sr) * kSH + k0 + sc];
    *(v8bf*)&As[buf][sr * kLdA + sc] = *(const v8bf*)(src);
    *(v8bf*)&As[buf][sr * kLdA + sc + 8] = *(const v8bf*)(src + 8);
  };

  stageA(0, 0);
  v8f acc[4] = {{}, {}, {}, {}};
  const float* wcol = wdn + (size_t)(half * 16) * kH + ncol + (lane & 15);
  for (int k0 = 0, it = 0; k0 < kSH; k0 += 32, ++it) {
    __syncthreads();
    if (k0 + 32 < kSH) {
      stageA((it + 1) & 1, k0 + 32);
      __builtin_prefetch(wcol + (size_t)32 * kH, 0, 0);
    }
    const v16bf b = frag_b_gmem(wcol, kH);
    mma_step4(&As[it & 1][0], lane, b, acc);
    wcol += (size_t)32 * kH;
  }

  const int c = lane & 15;
  const int rb = half * 8;
#pragma unroll
  for (int m = 0; m < 4; ++m)
#pragma unroll
    for (int r = 0; r < 8; ++r)
      y[(size_t)(t0 + m * 16 + rb + r) * kH + ncol + c] = acc[m][r];
}

// -------------------------------------------------- routed grouped FFN1 ----
__global__ void __launch_bounds__(256) moe_ffn1_kernel(
    const float* __restrict__ x, const float* __restrict__ w_gate,
    const float* __restrict__ w_up, const int* __restrict__ counts,
    const int* __restrict__ tok_ids, bf16* __restrict__ hbuf) {
  const int tilesPerE = kCap / kBM;   // 24
  const int e = blockIdx.y / tilesPerE;
  const int slot0 = (blockIdx.y % tilesPerE) * kBM;
  int cnt = counts[e];
  if (cnt > kCap) cnt = kCap;
  if (slot0 >= cnt) return;

  const int wave = threadIdx.x >> 5;
  const int lane = threadIdx.x & 31;
  const int half = lane >> 4;
  const int nsub = wave & 3;
  const int ncol = blockIdx.x * 64 + nsub * 16;
  const float* __restrict__ W =
      ((wave < 4) ? w_gate : w_up) + (size_t)e * kH * kI;

  __shared__ __align__(16) bf16 As[2][kBM * kLdA];
  __shared__ float Us[4][kBM][17];
  __shared__ int toks[kBM];

  if (threadIdx.x < kBM) {
    const int s = slot0 + threadIdx.x;
    toks[threadIdx.x] = (s < cnt) ? tok_ids[e * kCap + s] : 0;
  }
  __syncthreads();

  const int sr = threadIdx.x >> 2;
  const int sc = (threadIdx.x & 3) * 8;
  auto stageA = [&](int buf, int k0) {
    const float* src = &x[(size_t)toks[sr] * kH + k0 + sc];
    const float4 f0 = ((const float4*)src)[0];
    const float4 f1 = ((const float4*)src)[1];
    *(v8bf*)&As[buf][sr * kLdA + sc] = pack8(f0, f1);
  };

  stageA(0, 0);
  v8f acc[4] = {{}, {}, {}, {}};
  const float* wcol = W + (size_t)(half * 16) * kI + ncol + (lane & 15);
  for (int k0 = 0, it = 0; k0 < kH; k0 += 32, ++it) {
    __syncthreads();
    if (k0 + 32 < kH) {
      stageA((it + 1) & 1, k0 + 32);
      __builtin_prefetch(wcol + (size_t)32 * kI, 0, 0);
    }
    const v16bf b = frag_b_gmem(wcol, kI);
    mma_step4(&As[it & 1][0], lane, b, acc);
    wcol += (size_t)32 * kI;
  }

  const int c = lane & 15;
  const int rb = half * 8;
  if (wave >= 4) {
#pragma unroll
    for (int m = 0; m < 4; ++m)
#pragma unroll
      for (int r = 0; r < 8; ++r) Us[nsub][m * 16 + rb + r][c] = acc[m][r];
  }
  __syncthreads();
  if (wave < 4) {
    const size_t base = (size_t)e * kCap + slot0;
#pragma unroll
    for (int m = 0; m < 4; ++m)
#pragma unroll
      for (int r = 0; r < 8; ++r) {
        const float g = acc[m][r];
        const float u = Us[nsub][m * 16 + rb + r][c];
        hbuf[(base + m * 16 + rb + r) * kI + ncol + c] =
            (bf16)(silu_f(g) * u);
      }
  }
}

// ------------------------------------- routed FFN2 + weighted combine ----
__global__ void __launch_bounds__(128) moe_ffn2_kernel(
    const bf16* __restrict__ hbuf, const float* __restrict__ w_down,
    const int* __restrict__ counts, const int* __restrict__ tok_ids,
    const float* __restrict__ wslot, float* __restrict__ y) {
  const int tilesPerE = kCap / kBM;
  const int e = blockIdx.y / tilesPerE;
  const int slot0 = (blockIdx.y % tilesPerE) * kBM;
  int cnt = counts[e];
  if (cnt > kCap) cnt = kCap;
  if (slot0 >= cnt) return;

  const int wave = threadIdx.x >> 5;
  const int lane = threadIdx.x & 31;
  const int half = lane >> 4;
  const int ncol = blockIdx.x * 64 + wave * 16;
  const float* __restrict__ Wd = w_down + (size_t)e * kI * kH;
  const bf16* __restrict__ hrow = hbuf + ((size_t)e * kCap + slot0) * kI;

  __shared__ __align__(16) bf16 As[2][kBM * kLdA];
  __shared__ int toks[kBM];
  __shared__ float tws[kBM];
  if (threadIdx.x < kBM) {
    const int s = slot0 + threadIdx.x;
    const bool valid = s < cnt;
    toks[threadIdx.x] = valid ? tok_ids[e * kCap + s] : 0;
    tws[threadIdx.x] = valid ? wslot[e * kCap + s] : 0.0f;
  }

  const int sr = threadIdx.x >> 1;
  const int sc = (threadIdx.x & 1) * 16;
  auto stageA = [&](int buf, int k0) {
    const bf16* src = &hrow[(size_t)sr * kI + k0 + sc];
    *(v8bf*)&As[buf][sr * kLdA + sc] = *(const v8bf*)(src);
    *(v8bf*)&As[buf][sr * kLdA + sc + 8] = *(const v8bf*)(src + 8);
  };

  stageA(0, 0);
  v8f acc[4] = {{}, {}, {}, {}};
  const float* wcol = Wd + (size_t)(half * 16) * kH + ncol + (lane & 15);
  for (int k0 = 0, it = 0; k0 < kI; k0 += 32, ++it) {
    __syncthreads();
    if (k0 + 32 < kI) {
      stageA((it + 1) & 1, k0 + 32);
      __builtin_prefetch(wcol + (size_t)32 * kH, 0, 0);
    }
    const v16bf b = frag_b_gmem(wcol, kH);
    mma_step4(&As[it & 1][0], lane, b, acc);
    wcol += (size_t)32 * kH;
  }

  const int c = lane & 15;
  const int rb = half * 8;
#pragma unroll
  for (int m = 0; m < 4; ++m)
#pragma unroll
    for (int r = 0; r < 8; ++r) {
      const int row = m * 16 + rb + r;
      const float w = tws[row];
      if (w != 0.0f)
        atomicAdd(&y[(size_t)toks[row] * kH + ncol + c], acc[m][r] * w);
    }
}

// ------------------------------------------------------------ launcher ----
extern "C" void kernel_launch(void* const* d_in, const int* in_sizes, int n_in,
                              void* d_out, int out_size, void* d_ws, size_t ws_size,
                              hipStream_t stream) {
  (void)in_sizes; (void)n_in; (void)out_size; (void)ws_size;
  const float* x         = (const float*)d_in[0];
  const float* gate_w    = (const float*)d_in[1];
  const float* gate_bias = (const float*)d_in[2];
  const float* w_gate    = (const float*)d_in[3];
  const float* w_up      = (const float*)d_in[4];
  const float* w_down    = (const float*)d_in[5];
  const float* ws_gate   = (const float*)d_in[6];
  const float* ws_up     = (const float*)d_in[7];
  const float* ws_down   = (const float*)d_in[8];
  float* y = (float*)d_out;

  char* ws = (char*)d_ws;
  size_t off = 0;
  int* counts = (int*)(ws + off);    off += 256;
  int* tok_ids = (int*)(ws + off);   off += (size_t)kE * kCap * sizeof(int);
  float* wslot = (float*)(ws + off); off += (size_t)kE * kCap * sizeof(float);
  bf16* hbuf = (bf16*)(ws + off);    off += (size_t)kE * kCap * kI * sizeof(bf16);
  bf16* hs = (bf16*)(ws + off);      off += (size_t)kT * kSH * sizeof(bf16);
  // total scratch: ~81 MB

  init_kernel<<<dim3(1), dim3(32), 0, stream>>>(counts);
  gate_kernel<<<dim3(kT), dim3(64), 0, stream>>>(x, gate_w, gate_bias, counts,
                                                 tok_ids, wslot);
  shared_ffn1_kernel<<<dim3(kSH / 64, kT / kBM), dim3(256), 0, stream>>>(
      x, ws_gate, ws_up, hs);
  shared_ffn2_kernel<<<dim3(kH / 64, kT / kBM), dim3(128), 0, stream>>>(
      hs, ws_down, y);
  moe_ffn1_kernel<<<dim3(kI / 64, kE * (kCap / kBM)), dim3(256), 0, stream>>>(
      x, w_gate, w_up, counts, tok_ids, hbuf);
  moe_ffn2_kernel<<<dim3(kH / 64, kE * (kCap / kBM)), dim3(128), 0, stream>>>(
      hbuf, w_down, counts, tok_ids, wslot, y);
}
